// VisionTransformer_3985729651526
// MI455X (gfx1250) — matrix operand order repaired and checked by reading, not compiled, not run on hardware
//
#include <hip/hip_runtime.h>
#include <hip/hip_bf16.h>
#include <math.h>

// ---------------- constants ----------------
#define BATCH     32
#define DEPTH     12
#define EMB       768
#define NHEAD     12
#define HDIM      64
#define IMG       224
#define PS        16
#define GRID_     14
#define NTOK      197            // 196 patches + cls
#define NTOK_P    208            // padded to 16 multiple (13 tiles)
#define NTOK_K    224            // padded to 32 multiple for att@V K-dim
#define TOKROWS   (BATCH * NTOK)     // 6304 = 32 * 197
#define PATROWS   (BATCH * 196)      // 6272 = 32 * 196
#define QKVC      (3 * EMB)          // 2304
#define FFN       (4 * EMB)          // 3072
#define NRD       732                // (2*14-1)^2 + 3
#define NCLS      1000
#define NCLS_P    1024

typedef __attribute__((ext_vector_type(8)))  float   v8f;
typedef __attribute__((ext_vector_type(8)))  __bf16  v8bf;
typedef __attribute__((ext_vector_type(16))) __bf16  v16bf;

// ---------------- helpers ----------------
__device__ __forceinline__ __bf16 f2bf(float f) {
  unsigned u = __builtin_bit_cast(unsigned, f);
  unsigned r = u + 0x7FFFu + ((u >> 16) & 1u);
  unsigned short h = (unsigned short)(r >> 16);
  return __builtin_bit_cast(__bf16, h);
}

__device__ __forceinline__ float gelu_exact(float x) {
  return 0.5f * x * (1.0f + erff(x * 0.7071067811865476f));
}

// A fragment (16x32 bf16, M x K). lane&15 = row; lane>>4 selects K half.
// element e -> K = kBase + (e>=8 ? 16 : 0) + kh*8 + (e&7)
__device__ __forceinline__ v16bf load_a_frag(const __bf16* __restrict__ A, int lda,
                                             int mBase, int kBase, int lane) {
  int row = mBase + (lane & 15);
  int kh  = (lane >> 4) & 1;
  const __bf16* p = A + (size_t)row * lda + kBase + kh * 8;
  v8bf lo = *(const v8bf*)p;
  v8bf hi = *(const v8bf*)(p + 16);
  v16bf r;
#pragma unroll
  for (int i = 0; i < 8; ++i) { r[i] = lo[i]; r[8 + i] = hi[i]; }
  return r;
}

// B fragment (32x16 bf16, K x N) from B^T stored row-major [N, K].
// lane&15 = col; element e -> K = kBase + (e>=8 ? 16 : 0) + kh*8 + (e&7)
__device__ __forceinline__ v16bf load_b_frag(const __bf16* __restrict__ Bt, int ldb,
                                             int nBase, int kBase, int lane) {
  int col = nBase + (lane & 15);
  int kh  = (lane >> 4) & 1;
  const __bf16* p = Bt + (size_t)col * ldb + kBase + kh * 8;
  v8bf lo = *(const v8bf*)p;
  v8bf hi = *(const v8bf*)(p + 16);
  v16bf r;
#pragma unroll
  for (int i = 0; i < 8; ++i) { r[i] = lo[i]; r[8 + i] = hi[i]; }
  return r;
}

__device__ __forceinline__ v8f wmma_bf16(v16bf a, v16bf b, v8f c) {
  return __builtin_amdgcn_wmma_f32_16x16x32_bf16(false, a, false, b, (short)0, c,
                                                 false, false);
}

// ---------------- generic bf16 GEMM: C = A @ Bt^T ----------------
// A: [M,K] bf16 row-major (M must be a multiple of 32).
// Bt: [N,K] bf16 row-major (i.e. B transposed), N multiple of 64.
// 256 threads = 8 waves; each wave computes a 32(M) x 64(N) tile:
// 2 A-fragments + 4 B-fragments -> 8 WMMAs per K-step of 32.
// EPI 0: outF = acc + bias           (f32)
// EPI 1: outB = bf16(gelu(acc+bias))
// EPI 2: outF += acc + bias          (residual accumulate, in place)
template <int EPI>
__global__ __launch_bounds__(256) void gemm_bf16_kernel(
    const __bf16* __restrict__ A, const __bf16* __restrict__ Bt,
    int M, int N, int K, int Nstore,
    const float* __restrict__ bias,
    float* __restrict__ outF, int ldo, __bf16* __restrict__ outB) {
  int lane  = threadIdx.x & 31;
  int wave  = threadIdx.x >> 5;
  int mBase = (blockIdx.x * 8 + wave) * 32;
  if (mBase >= M) return;
  int nBase = blockIdx.y * 64;

  v8f acc[2][4] = {};
  for (int k = 0; k < K; k += 32) {
    v16bf a0 = load_a_frag(A, K, mBase,      k, lane);
    v16bf a1 = load_a_frag(A, K, mBase + 16, k, lane);
    v16bf b0 = load_b_frag(Bt, K, nBase +  0, k, lane);
    v16bf b1 = load_b_frag(Bt, K, nBase + 16, k, lane);
    v16bf b2 = load_b_frag(Bt, K, nBase + 32, k, lane);
    v16bf b3 = load_b_frag(Bt, K, nBase + 48, k, lane);
    if (k + 32 < K) {
      __builtin_prefetch(A  + (size_t)(mBase + (lane & 31)) * K + k + 32, 0, 3);
      __builtin_prefetch(Bt + (size_t)(nBase + (lane & 31)) * K + k + 32, 0, 3);
    }
    acc[0][0] = wmma_bf16(a0, b0, acc[0][0]);
    acc[0][1] = wmma_bf16(a0, b1, acc[0][1]);
    acc[0][2] = wmma_bf16(a0, b2, acc[0][2]);
    acc[0][3] = wmma_bf16(a0, b3, acc[0][3]);
    acc[1][0] = wmma_bf16(a1, b0, acc[1][0]);
    acc[1][1] = wmma_bf16(a1, b1, acc[1][1]);
    acc[1][2] = wmma_bf16(a1, b2, acc[1][2]);
    acc[1][3] = wmma_bf16(a1, b3, acc[1][3]);
  }

  int kh   = (lane >> 4) & 1;
  int col0 = nBase + (lane & 15);
#pragma unroll
  for (int ms = 0; ms < 2; ++ms) {
#pragma unroll
    for (int t = 0; t < 4; ++t) {
      int col = col0 + t * 16;
      if (col >= Nstore) continue;
      float bv = bias ? bias[col] : 0.0f;
#pragma unroll
      for (int r = 0; r < 8; ++r) {
        int row = mBase + ms * 16 + r + kh * 8;
        if (row >= M) continue;
        size_t o = (size_t)row * ldo + col;
        float v = acc[ms][t][r] + bv;
        if (EPI == 0)      outF[o] = v;
        else if (EPI == 1) outB[o] = f2bf(gelu_exact(v));
        else               outF[o] = outF[o] + v;
      }
    }
  }
}

// ---------------- attention: scores = Q @ K^T ----------------
// q,k: [B*H, 208, 64] bf16. scores: [B*H, 208, 208] f32. One wave per 16x16 tile.
__global__ __launch_bounds__(32) void attn_scores_kernel(
    const __bf16* __restrict__ q, const __bf16* __restrict__ k,
    float* __restrict__ scores) {
  int lane  = threadIdx.x & 31;
  int bh    = blockIdx.z;
  int mBase = blockIdx.x * 16;
  int nBase = blockIdx.y * 16;
  const __bf16* Aq = q + (size_t)bh * NTOK_P * HDIM;
  const __bf16* Bk = k + (size_t)bh * NTOK_P * HDIM;
  v8f acc = {};
#pragma unroll
  for (int kk = 0; kk < HDIM; kk += 32) {
    v16bf a = load_a_frag(Aq, HDIM, mBase, kk, lane);
    v16bf b = load_b_frag(Bk, HDIM, nBase, kk, lane);
    acc = wmma_bf16(a, b, acc);
  }
  int kh  = (lane >> 4) & 1;
  int col = nBase + (lane & 15);
  float* srow = scores + (size_t)bh * NTOK_P * NTOK_P;
#pragma unroll
  for (int r = 0; r < 8; ++r)
    srow[(size_t)(mBase + r + kh * 8) * NTOK_P + col] = acc[r];
}

// ---------------- attention: O = att @ V ----------------
// att: [B*H, 208, 224] bf16 (zero padded cols). vt: [B*H, 64, 224] bf16
// (V transposed, zero padded). Writes O into [TOKROWS, EMB] bf16 at col h*64.
__global__ __launch_bounds__(32) void attn_v_kernel(
    const __bf16* __restrict__ att, const __bf16* __restrict__ vt,
    __bf16* __restrict__ obuf) {
  int lane  = threadIdx.x & 31;
  int bh    = blockIdx.z;
  int b     = bh / NHEAD, h = bh % NHEAD;
  int mBase = blockIdx.x * 16;
  int nBase = blockIdx.y * 16;
  const __bf16* Aa = att + (size_t)bh * NTOK_P * NTOK_K;
  const __bf16* Bv = vt  + (size_t)bh * HDIM * NTOK_K;
  v8f acc = {};
  for (int kk = 0; kk < NTOK_K; kk += 32) {
    v16bf a   = load_a_frag(Aa, NTOK_K, mBase, kk, lane);
    v16bf bfr = load_b_frag(Bv, NTOK_K, nBase, kk, lane);
    acc = wmma_bf16(a, bfr, acc);
  }
  int kh  = (lane >> 4) & 1;
  int col = h * HDIM + nBase + (lane & 15);
#pragma unroll
  for (int r = 0; r < 8; ++r) {
    int tok = mBase + r + kh * 8;
    if (tok < NTOK)
      obuf[(size_t)(b * NTOK + tok) * EMB + col] = f2bf(acc[r]);
  }
}

// ---------------- softmax with BEiT relative position bias ----------------
__device__ __forceinline__ int rel_idx(int m, int n) {
  if (m == 0 && n == 0) return NRD - 1;  // 731
  if (m == 0)           return NRD - 3;  // 729
  if (n == 0)           return NRD - 2;  // 730
  int a = m - 1, c = n - 1;
  int dr = a / GRID_ - c / GRID_ + (GRID_ - 1);
  int dc = a % GRID_ - c % GRID_ + (GRID_ - 1);
  return dr * (2 * GRID_ - 1) + dc;
}

__global__ __launch_bounds__(32) void softmax_bias_kernel(
    const float* __restrict__ scores, const float* __restrict__ rel_table_d,
    __bf16* __restrict__ att) {
  int m    = blockIdx.x % NTOK;
  int bh   = blockIdx.x / NTOK;
  int h    = bh % NHEAD;
  int lane = threadIdx.x;
  const float* srow = scores + ((size_t)bh * NTOK_P + m) * NTOK_P;
  __bf16* arow = att + ((size_t)bh * NTOK_P + m) * NTOK_K;

  float vals[7];
  float mx = -3.0e38f;
#pragma unroll
  for (int i = 0; i < 7; ++i) {
    int n = lane + i * 32;
    float v = -3.0e38f;
    if (n < NTOK) v = srow[n] + rel_table_d[rel_idx(m, n) * NHEAD + h];
    vals[i] = v;
    mx = fmaxf(mx, v);
  }
#pragma unroll
  for (int off = 16; off > 0; off >>= 1) mx = fmaxf(mx, __shfl_xor(mx, off, 32));
  float s = 0.0f;
#pragma unroll
  for (int i = 0; i < 7; ++i) {
    int n = lane + i * 32;
    float e = (n < NTOK) ? __expf(vals[i] - mx) : 0.0f;
    vals[i] = e;
    s += e;
  }
#pragma unroll
  for (int off = 16; off > 0; off >>= 1) s += __shfl_xor(s, off, 32);
  float inv = 1.0f / s;
#pragma unroll
  for (int i = 0; i < 7; ++i) {
    int n = lane + i * 32;
    if (n < NTOK_K) arow[n] = f2bf((n < NTOK) ? vals[i] * inv : 0.0f);
  }
}

// ---------------- LayerNorm over E=768 (block = 256 = 8 waves) ----------------
__global__ __launch_bounds__(256) void layernorm_kernel(
    const float* __restrict__ x, const float* __restrict__ w,
    const float* __restrict__ b, __bf16* __restrict__ outB) {
  int row = blockIdx.x;
  const float* xr = x + (size_t)row * EMB;
  int lane = threadIdx.x & 31, wave = threadIdx.x >> 5;
  float s = 0.0f, s2 = 0.0f;
  for (int i = threadIdx.x; i < EMB; i += 256) {
    float v = xr[i];
    s += v; s2 += v * v;
  }
#pragma unroll
  for (int off = 16; off > 0; off >>= 1) {
    s  += __shfl_xor(s, off, 32);
    s2 += __shfl_xor(s2, off, 32);
  }
  __shared__ float ws1[8], ws2[8];
  if (lane == 0) { ws1[wave] = s; ws2[wave] = s2; }
  __syncthreads();
  float ts = 0.0f, ts2 = 0.0f;
#pragma unroll
  for (int j = 0; j < 8; ++j) { ts += ws1[j]; ts2 += ws2[j]; }
  float mean = ts * (1.0f / EMB);
  float var  = ts2 * (1.0f / EMB) - mean * mean;
  float rstd = rsqrtf(var + 1e-5f);
  for (int i = threadIdx.x; i < EMB; i += 256) {
    float v = (xr[i] - mean) * rstd * w[i] + b[i];
    outB[(size_t)row * EMB + i] = f2bf(v);
  }
}

// ---------------- misc elementwise kernels ----------------
__global__ void im2col_kernel(const float* __restrict__ x, __bf16* __restrict__ P) {
  size_t idx = (size_t)blockIdx.x * blockDim.x + threadIdx.x;
  if (idx >= (size_t)PATROWS * EMB) return;
  int row = (int)(idx / EMB), col = (int)(idx % EMB);
  int b = row / 196, p = row % 196;
  int gy = p / GRID_, gx = p % GRID_;
  int c = col / (PS * PS), rem = col % (PS * PS);
  int py = rem / PS, px = rem % PS;
  float v = x[(((size_t)b * 3 + c) * IMG + gy * PS + py) * IMG + gx * PS + px];
  P[idx] = f2bf(v);
}

__global__ void f32_to_bf16_kernel(const float* __restrict__ src,
                                   __bf16* __restrict__ dst, int n_src, int n_tot) {
  int idx = blockIdx.x * blockDim.x + threadIdx.x;
  if (idx >= n_tot) return;
  dst[idx] = (idx < n_src) ? f2bf(src[idx]) : f2bf(0.0f);
}

__global__ void build_qkvb_kernel(const float* __restrict__ qb,
                                  const float* __restrict__ vb,
                                  float* __restrict__ qkvb) {
  int i = blockIdx.x * blockDim.x + threadIdx.x;
  if (i >= QKVC) return;
  qkvb[i] = (i < EMB) ? qb[i] : ((i < 2 * EMB) ? 0.0f : vb[i - 2 * EMB]);
}

// qkvF [TOKROWS, 2304] f32 (bias included) -> q/k [B*H,208,64] bf16 (q scaled),
// vt [B*H,64,224] bf16 transposed, padding zeroed.
__global__ void qkv_split_kernel(const float* __restrict__ qkvF,
                                 __bf16* __restrict__ q, __bf16* __restrict__ k,
                                 __bf16* __restrict__ vt) {
  size_t idx = (size_t)blockIdx.x * blockDim.x + threadIdx.x;
  if (idx >= (size_t)BATCH * NHEAD * NTOK_K * HDIM) return;
  int hd  = (int)(idx % HDIM);
  int tok = (int)((idx / HDIM) % NTOK_K);
  int bh  = (int)(idx / ((size_t)HDIM * NTOK_K));
  int b = bh / NHEAD, h = bh % NHEAD;
  float qv = 0.0f, kv = 0.0f, vv = 0.0f;
  if (tok < NTOK) {
    const float* base = qkvF + (size_t)(b * NTOK + tok) * QKVC + h * HDIM + hd;
    qv = base[0] * 0.125f;   // HD^-0.5
    kv = base[EMB];
    vv = base[2 * EMB];
  }
  if (tok < NTOK_P) {
    q[((size_t)bh * NTOK_P + tok) * HDIM + hd] = f2bf(qv);
    k[((size_t)bh * NTOK_P + tok) * HDIM + hd] = f2bf(kv);
  }
  vt[((size_t)bh * HDIM + hd) * NTOK_K + tok] = f2bf(vv);
}

__global__ void assemble_tok_kernel(const float* __restrict__ Pout,
                                    const float* __restrict__ cls,
                                    float* __restrict__ tok) {
  size_t idx = (size_t)blockIdx.x * blockDim.x + threadIdx.x;
  if (idx >= (size_t)TOKROWS * EMB) return;
  int row = (int)(idx / EMB), e = (int)(idx % EMB);
  int b = row / NTOK, t = row % NTOK;
  tok[idx] = (t == 0) ? cls[e] : Pout[(size_t)(b * 196 + t - 1) * EMB + e];
}

__global__ __launch_bounds__(256) void meanpool_kernel(const float* __restrict__ tok,
                                                       float* __restrict__ feat) {
  int b = blockIdx.x;
  for (int e = threadIdx.x; e < EMB; e += 256) {
    float s = 0.0f;
    for (int t = 1; t < NTOK; ++t) s += tok[(size_t)(b * NTOK + t) * EMB + e];
    feat[(size_t)b * EMB + e] = s * (1.0f / 196.0f);
  }
}

// ---------------- host launch ----------------
static inline dim3 gemm_grid(int M, int N) {
  return dim3((unsigned)((M / 32 + 7) / 8), (unsigned)(N / 64));
}

extern "C" void kernel_launch(void* const* d_in, const int* in_sizes, int n_in,
                              void* d_out, int out_size, void* d_ws, size_t ws_size,
                              hipStream_t stream) {
  const float* x        = (const float*)d_in[0];
  const float* patch_w  = (const float*)d_in[1];
  const float* patch_b  = (const float*)d_in[2];
  const float* cls_tok  = (const float*)d_in[3];
  const float* ln1_w    = (const float*)d_in[4];
  const float* ln1_b    = (const float*)d_in[5];
  const float* qkv_w    = (const float*)d_in[6];
  const float* q_bias   = (const float*)d_in[7];
  const float* v_bias   = (const float*)d_in[8];
  const float* rel_tab  = (const float*)d_in[9];
  const float* proj_w   = (const float*)d_in[10];
  const float* proj_b   = (const float*)d_in[11];
  const float* ln2_w    = (const float*)d_in[12];
  const float* ln2_b    = (const float*)d_in[13];
  const float* fc1_w    = (const float*)d_in[14];
  const float* fc1_b    = (const float*)d_in[15];
  const float* fc2_w    = (const float*)d_in[16];
  const float* fc2_b    = (const float*)d_in[17];
  const float* fcn_w    = (const float*)d_in[18];
  const float* fcn_b    = (const float*)d_in[19];
  const float* head_w   = (const float*)d_in[20];
  const float* head_b   = (const float*)d_in[21];
  float* out = (float*)d_out;

  // ---- carve workspace (256B aligned) ----
  size_t off = 0;
  auto alloc = [&](size_t bytes) -> char* {
    char* p = (char*)d_ws + off;
    off += (bytes + 255) & ~(size_t)255;
    return p;
  };
  float*  tokF    = (float*) alloc((size_t)TOKROWS * EMB * 4);
  __bf16* y_bf    = (__bf16*)alloc((size_t)TOKROWS * EMB * 2);
  float*  qkvF    = (float*) alloc((size_t)TOKROWS * QKVC * 4);   // also patch-GEMM out
  __bf16* q_bf    = (__bf16*)alloc((size_t)BATCH * NHEAD * NTOK_P * HDIM * 2);
  __bf16* k_bf    = (__bf16*)alloc((size_t)BATCH * NHEAD * NTOK_P * HDIM * 2);
  __bf16* vt_bf   = (__bf16*)alloc((size_t)BATCH * NHEAD * HDIM * NTOK_K * 2);
  float*  scoresF = (float*) alloc((size_t)BATCH * NHEAD * NTOK_P * NTOK_P * 4);
  __bf16* att_bf  = (__bf16*)alloc((size_t)BATCH * NHEAD * NTOK_P * NTOK_K * 2);
  __bf16* o_bf    = (__bf16*)alloc((size_t)TOKROWS * EMB * 2);
  __bf16* h1_bf   = (__bf16*)alloc((size_t)TOKROWS * FFN * 2);    // also im2col P
  __bf16* wqkv_bf = (__bf16*)alloc((size_t)QKVC * EMB * 2);
  __bf16* wproj_bf= (__bf16*)alloc((size_t)EMB * EMB * 2);
  __bf16* wfc1_bf = (__bf16*)alloc((size_t)FFN * EMB * 2);
  __bf16* wfc2_bf = (__bf16*)alloc((size_t)EMB * FFN * 2);
  __bf16* wpat_bf = (__bf16*)alloc((size_t)EMB * EMB * 2);
  __bf16* whead_bf= (__bf16*)alloc((size_t)NCLS_P * EMB * 2);
  float*  qkvb    = (float*) alloc((size_t)QKVC * 4);
  float*  featF   = (float*) alloc((size_t)BATCH * EMB * 4);
  __bf16* feat_bf = (__bf16*)alloc((size_t)BATCH * EMB * 2);
  __bf16* P_bf    = h1_bf;      // alias: im2col buffer lives where h1 will
  float*  PoutF   = qkvF;       // alias: patch-GEMM f32 output

  const int T256 = 256;
  auto nb = [](size_t n) { return (unsigned)((n + 255) / 256); };

  // ---- patch embedding ----
  f32_to_bf16_kernel<<<nb((size_t)EMB * EMB), T256, 0, stream>>>(
      patch_w, wpat_bf, EMB * EMB, EMB * EMB);
  im2col_kernel<<<nb((size_t)PATROWS * EMB), T256, 0, stream>>>(x, P_bf);
  gemm_bf16_kernel<0><<<gemm_grid(PATROWS, EMB), T256, 0, stream>>>(
      P_bf, wpat_bf, PATROWS, EMB, EMB, EMB, patch_b, PoutF, EMB, nullptr);
  assemble_tok_kernel<<<nb((size_t)TOKROWS * EMB), T256, 0, stream>>>(
      PoutF, cls_tok, tokF);

  // ---- transformer blocks ----
  for (int d = 0; d < DEPTH; ++d) {
    f32_to_bf16_kernel<<<nb((size_t)QKVC * EMB), T256, 0, stream>>>(
        qkv_w + (size_t)d * QKVC * EMB, wqkv_bf, QKVC * EMB, QKVC * EMB);
    f32_to_bf16_kernel<<<nb((size_t)EMB * EMB), T256, 0, stream>>>(
        proj_w + (size_t)d * EMB * EMB, wproj_bf, EMB * EMB, EMB * EMB);
    f32_to_bf16_kernel<<<nb((size_t)FFN * EMB), T256, 0, stream>>>(
        fc1_w + (size_t)d * FFN * EMB, wfc1_bf, FFN * EMB, FFN * EMB);
    f32_to_bf16_kernel<<<nb((size_t)EMB * FFN), T256, 0, stream>>>(
        fc2_w + (size_t)d * EMB * FFN, wfc2_bf, EMB * FFN, EMB * FFN);
    build_qkvb_kernel<<<nb(QKVC), T256, 0, stream>>>(
        q_bias + (size_t)d * EMB, v_bias + (size_t)d * EMB, qkvb);

    // attention
    layernorm_kernel<<<TOKROWS, T256, 0, stream>>>(
        tokF, ln1_w + (size_t)d * EMB, ln1_b + (size_t)d * EMB, y_bf);
    gemm_bf16_kernel<0><<<gemm_grid(TOKROWS, QKVC), T256, 0, stream>>>(
        y_bf, wqkv_bf, TOKROWS, QKVC, EMB, QKVC, qkvb, qkvF, QKVC, nullptr);
    qkv_split_kernel<<<nb((size_t)BATCH * NHEAD * NTOK_K * HDIM), T256, 0, stream>>>(
        qkvF, q_bf, k_bf, vt_bf);
    attn_scores_kernel<<<dim3(13, 13, BATCH * NHEAD), 32, 0, stream>>>(
        q_bf, k_bf, scoresF);
    softmax_bias_kernel<<<BATCH * NHEAD * NTOK, 32, 0, stream>>>(
        scoresF, rel_tab + (size_t)d * NRD * NHEAD, att_bf);
    attn_v_kernel<<<dim3(13, HDIM / 16, BATCH * NHEAD), 32, 0, stream>>>(
        att_bf, vt_bf, o_bf);
    gemm_bf16_kernel<2><<<gemm_grid(TOKROWS, EMB), T256, 0, stream>>>(
        o_bf, wproj_bf, TOKROWS, EMB, EMB, EMB,
        proj_b + (size_t)d * EMB, tokF, EMB, nullptr);

    // MLP
    layernorm_kernel<<<TOKROWS, T256, 0, stream>>>(
        tokF, ln2_w + (size_t)d * EMB, ln2_b + (size_t)d * EMB, y_bf);
    gemm_bf16_kernel<1><<<gemm_grid(TOKROWS, FFN), T256, 0, stream>>>(
        y_bf, wfc1_bf, TOKROWS, FFN, EMB, FFN,
        fc1_b + (size_t)d * FFN, nullptr, FFN, h1_bf);
    gemm_bf16_kernel<2><<<gemm_grid(TOKROWS, EMB), T256, 0, stream>>>(
        h1_bf, wfc2_bf, TOKROWS, EMB, FFN, EMB,
        fc2_b + (size_t)d * EMB, tokF, EMB, nullptr);
  }

  // ---- head: mean-pool patch tokens, final LN, classifier ----
  meanpool_kernel<<<BATCH, T256, 0, stream>>>(tokF, featF);
  layernorm_kernel<<<BATCH, T256, 0, stream>>>(featF, fcn_w, fcn_b, feat_bf);
  f32_to_bf16_kernel<<<nb((size_t)NCLS_P * EMB), T256, 0, stream>>>(
      head_w, whead_bf, NCLS * EMB, NCLS_P * EMB);
  gemm_bf16_kernel<0><<<gemm_grid(BATCH, NCLS_P), T256, 0, stream>>>(
      feat_bf, whead_bf, BATCH, NCLS_P, EMB, NCLS, head_b, out, NCLS, nullptr);
}
